// S4Block_26903675142800
// MI455X (gfx1250) — matrix operand order repaired and checked by reading, not compiled, not run on hardware
//
#include <hip/hip_runtime.h>
#include <hip/hip_bf16.h>

#define DIM 512
#define NST 64
#define BATCH 16
#define STEPS 4096
#define ROWS (BATCH * STEPS)   // 65536

typedef __attribute__((ext_vector_type(16))) __bf16 v16bf;
typedef __attribute__((ext_vector_type(8)))  float  v8f;
typedef __attribute__((ext_vector_type(2)))  float  v2f;

// ---------------------------------------------------------------------------
// K0: split mixer_w (fp32, 512x512, row e major, W[e][d]) into bf16 hi/lo
// ---------------------------------------------------------------------------
__global__ void wsplit_kernel(const float* __restrict__ w,
                              __bf16* __restrict__ whi,
                              __bf16* __restrict__ wlo) {
  int i = blockIdx.x * 256 + threadIdx.x;  // 262144 elements
  float v = w[i];
  __bf16 h = (__bf16)v;
  whi[i] = h;
  wlo[i] = (__bf16)(v - (float)h);
}

// ---------------------------------------------------------------------------
// K1: RMSNorm per (b,t) row of 512. 128 threads/row, float4 per thread.
// ---------------------------------------------------------------------------
__global__ void rmsnorm_kernel(const float* __restrict__ tokens,
                               const float* __restrict__ norm_w,
                               float* __restrict__ x) {
  int row = blockIdx.x;
  int tid = threadIdx.x;  // 0..127
  const float4* t = (const float4*)(tokens + (size_t)row * DIM);
  float4 v = t[tid];
  float ss = v.x * v.x + v.y * v.y + v.z * v.z + v.w * v.w;
  #pragma unroll
  for (int off = 16; off; off >>= 1) ss += __shfl_xor(ss, off, 32);
  __shared__ float red[4];
  if ((tid & 31) == 0) red[tid >> 5] = ss;
  __syncthreads();
  float tot = red[0] + red[1] + red[2] + red[3];
  float scale = rsqrtf(tot * (1.0f / (float)DIM) + 1e-4f);
  float4 w = ((const float4*)norm_w)[tid];
  float4 o;
  o.x = v.x * scale * w.x;
  o.y = v.y * scale * w.y;
  o.z = v.z * scale * w.z;
  o.w = v.w * scale * w.w;
  ((float4*)(x + (size_t)row * DIM))[tid] = o;
}

// ---------------------------------------------------------------------------
// K2: diagonal SSM scan, z-form + packed fp32 (v_pk_fma_f32).
//   Track z[n] = sum_{s<=t} A^(t-s) x[s]:   z = A*z + x      (1 FMA/state)
//   Output  y  = sum_n (C[d,n]*Bbar[d,n]) * z[n] + D*x       (1 FMA/state)
// 64 states split over 4 lanes (16 states = 8 v2f per lane).
// Block = 256 threads = 64 (b,d) groups. Grid = 16*512/64 = 128 blocks.
// ---------------------------------------------------------------------------
__global__ void scan_kernel(const float* __restrict__ x,
                            const float* __restrict__ A_log,
                            const float* __restrict__ log_dt,
                            const float* __restrict__ B_in,
                            const float* __restrict__ C_in,
                            const float* __restrict__ D_in,
                            float* __restrict__ y) {
  int tid = threadIdx.x;
  int grp = tid >> 2;        // 0..63: which (b,d) within block
  int sub = tid & 3;         // which 16-state slice
  int d = (blockIdx.x & 7) * 64 + grp;
  int b = blockIdx.x >> 3;

  float dt = log1pf(__expf(log_dt[d]));   // softplus
  float Dd = D_in[d];

  v2f A2[8], CB2[8], z2[8];
  #pragma unroll
  for (int j = 0; j < 8; j++) {
    int n = sub * 16 + j * 2;
    float a0 = __expf(-dt * __expf(A_log[d * NST + n]));
    float a1 = __expf(-dt * __expf(A_log[d * NST + n + 1]));
    A2[j]  = (v2f){a0, a1};
    CB2[j] = (v2f){C_in[d * NST + n]     * (dt * B_in[n]),
                   C_in[d * NST + n + 1] * (dt * B_in[n + 1])};
    z2[j]  = (v2f){0.f, 0.f};
  }

  const float* xp = x + ((size_t)b * STEPS) * DIM + d;
  float*       yp = y + ((size_t)b * STEPS) * DIM + d;

  for (int t = 0; t < STEPS; t++) {
    float xs = xp[(size_t)t * DIM];
    v2f xt2 = (v2f){xs, xs};
    v2f sa = (v2f){0.f, 0.f};
    v2f sb = (v2f){0.f, 0.f};
    #pragma unroll
    for (int j = 0; j < 8; j += 2) {
      z2[j]     = __builtin_elementwise_fma(A2[j],     z2[j],     xt2);
      z2[j + 1] = __builtin_elementwise_fma(A2[j + 1], z2[j + 1], xt2);
      sa = __builtin_elementwise_fma(CB2[j],     z2[j],     sa);
      sb = __builtin_elementwise_fma(CB2[j + 1], z2[j + 1], sb);
    }
    v2f st = sa + sb;
    float sum = st.x + st.y;
    sum += __shfl_xor(sum, 1, 32);
    sum += __shfl_xor(sum, 2, 32);
    if (sub == 0) yp[(size_t)t * DIM] = fmaf(Dd, xs, sum);
  }
}

// ---------------------------------------------------------------------------
// K3: mixer GEMM  out[r,e] = sum_d y[r,d]*W[e,d] + bias[e] + tokens[r,e]
// bf16 hi/lo split, 3x v_wmma_f32_16x16x32_bf16 per K-step (fp32-accurate).
// Four 16x16 tiles per wave. Block = 8 waves; wave w covers ntiles w*4..w*4+3,
// reusing its A fragment across the 4 N-tiles. Grid.x = ROWS/16 M-tiles.
// ---------------------------------------------------------------------------
__global__ void mixer_wmma_kernel(const float* __restrict__ y,
                                  const __bf16* __restrict__ whi,
                                  const __bf16* __restrict__ wlo,
                                  const float* __restrict__ bias,
                                  const float* __restrict__ tokens,
                                  float* __restrict__ out) {
  int mtile = blockIdx.x;                 // 0..4095
  int wave  = threadIdx.x >> 5;           // 0..7
  int lane  = threadIdx.x & 31;

  // A-fragment addressing (16-bit A 16x32, contiguous k order, same as B):
  // lanes 0-15 hold M=lane, K in {0..15}; lanes 16-31 same M, K in {16..31}.
  int mrow = lane & 15;
  int kb   = (lane >> 4) * 16;
  size_t arow = ((size_t)mtile * 16 + mrow) * DIM;

  v8f acc[4];
  #pragma unroll
  for (int q = 0; q < 4; q++) acc[q] = (v8f){0.f,0.f,0.f,0.f,0.f,0.f,0.f,0.f};

  for (int ks = 0; ks < DIM / 32; ks++) {
    int k0 = ks * 32;
    // Load + split A fragment from fp32 y
    v16bf ahi, alo;
    #pragma unroll
    for (int i = 0; i < 16; i++) {
      int p = i >> 1, w = i & 1;
      int k = p * 2 + w + kb;
      float v = y[arow + k0 + k];
      __bf16 h = (__bf16)v;
      ahi[i] = h;
      alo[i] = (__bf16)(v - (float)h);
    }
    #pragma unroll
    for (int q = 0; q < 4; q++) {
      int ntile = wave * 4 + q;
      // B-fragment (16-bit B 32x16): lanes 0-15 col=lane K=0..15,
      // lanes 16-31 same col K=16..31; element i -> K = khalf + i.
      int e  = ntile * 16 + (lane & 15);
      int kh = (lane >> 4) * 16;
      size_t boff = (size_t)e * DIM + k0 + kh;
      v16bf bhi, blo;
      #pragma unroll
      for (int i = 0; i < 16; i++) {
        bhi[i] = whi[boff + i];
        blo[i] = wlo[boff + i];
      }
      acc[q] = __builtin_amdgcn_wmma_f32_16x16x32_bf16(
          false, ahi, false, bhi, (short)0, acc[q], false, false);
      acc[q] = __builtin_amdgcn_wmma_f32_16x16x32_bf16(
          false, ahi, false, blo, (short)0, acc[q], false, false);
      acc[q] = __builtin_amdgcn_wmma_f32_16x16x32_bf16(
          false, alo, false, bhi, (short)0, acc[q], false, false);
    }
  }

  // C/D layout: VGPR r -> row = r + 8*(lane/16), col = lane%16
  int crow = (lane >> 4) * 8;
  int ccol = lane & 15;
  #pragma unroll
  for (int q = 0; q < 4; q++) {
    int e = (wave * 4 + q) * 16 + ccol;
    float bv = bias[e];
    #pragma unroll
    for (int r = 0; r < 8; r++) {
      size_t rr  = (size_t)mtile * 16 + crow + r;
      size_t idx = rr * DIM + e;
      out[idx] = acc[q][r] + bv + tokens[idx];
    }
  }
}

// ---------------------------------------------------------------------------
extern "C" void kernel_launch(void* const* d_in, const int* in_sizes, int n_in,
                              void* d_out, int out_size, void* d_ws, size_t ws_size,
                              hipStream_t stream) {
  const float* tokens  = (const float*)d_in[0];
  const float* norm_w  = (const float*)d_in[1];
  const float* A_log   = (const float*)d_in[2];
  const float* log_dt  = (const float*)d_in[3];
  const float* B_in    = (const float*)d_in[4];
  const float* C_in    = (const float*)d_in[5];
  const float* D_in    = (const float*)d_in[6];
  const float* mixer_w = (const float*)d_in[7];
  const float* mixer_b = (const float*)d_in[8];
  float* out = (float*)d_out;

  // Workspace layout: x (128MB) | y (128MB) | whi (512KB) | wlo (512KB)
  float*  x   = (float*)d_ws;
  float*  yv  = x + (size_t)ROWS * DIM;
  __bf16* whi = (__bf16*)(yv + (size_t)ROWS * DIM);
  __bf16* wlo = whi + (size_t)DIM * DIM;

  wsplit_kernel<<<(DIM * DIM) / 256, 256, 0, stream>>>(mixer_w, whi, wlo);
  rmsnorm_kernel<<<ROWS, 128, 0, stream>>>(tokens, norm_w, x);
  scan_kernel<<<(BATCH * DIM) / 64, 256, 0, stream>>>(x, A_log, log_dt, B_in,
                                                      C_in, D_in, yv);
  mixer_wmma_kernel<<<ROWS / 16, 256, 0, stream>>>(yv, whi, wlo, mixer_b,
                                                   tokens, out);
}
